// NodeMLPModel_15745350107780
// MI455X (gfx1250) — matrix operand order repaired and checked by reading, not compiled, not run on hardware
//
#include <hip/hip_runtime.h>
#include <hip/hip_bf16.h>

// MetaLayer edge/node MLP block for MI455X (gfx1250), bf16 WMMA pipeline.
//
// Pipeline (all on `stream`):
//   0. convert_w      : W1a/W1b/W2a/W2b f32 -> column-major bf16 in ws
//   1. zero           : stats + segment-sum buffer + counts
//   2. edge_pass1     : Z1 = [x[row]||ea] @ W1a + b1a   (WMMA bf16), accumulate
//                       per-channel sum/sumsq (LDS atomics -> global atomics)
//   3. bn_finalize    : scale = g*rsqrt(var+eps), shift = be - mean*scale
//   4. edge_pass2     : recompute Z1, h = selu(bn(Z1)), O = h @ W1b + b1b,
//                       atomic scatter-add into s[col], cnt[col]
//   5. node_pass1     : Z2 = [x||s/cnt||u[batch]] @ W2a + b2a, stats
//   6. bn_finalize
//   7. node_pass2     : recompute Z2, h = selu(bn(Z2)), out = h @ W2b + b2b
//
// Workspace requirement ~52 MB (s: N*128 f32 dominates).

typedef __attribute__((ext_vector_type(16))) __bf16 v16bf;
typedef __attribute__((ext_vector_type(8)))  __bf16 v8bf;
typedef __attribute__((ext_vector_type(8)))  float  v8f;

#define XD 128
#define ED 64
#define UD 128
#define HD 128
#define K1 (XD + ED)        // 192
#define K2 (XD + HD + UD)   // 384

// workspace byte offsets
#define OFF_W1AT   0
#define OFF_W1BT   49152
#define OFF_W2AT   81920
#define OFF_W2BT   180224
#define OFF_STATS1 212992   // 256 f32 (sum, sumsq)
#define OFF_BN1    214016   // 256 f32 (scale, shift)
#define OFF_STATS2 215040
#define OFF_BN2    216064
#define OFF_S      262144   // N*128 f32, then cnt: N f32

__device__ __forceinline__ float selu_f(float v) {
    const float lam = 1.0507009873554805f;
    const float alp = 1.6732632423543772f;
    return v > 0.0f ? lam * v : lam * alp * (expf(v) - 1.0f);
}

// A fragment (16xK tile, bf16, row-major in LDS with stride strideK).
// ISA layout: lanes 0-15 -> M=lane, K = {kb..kb+7, kb+16..kb+23};
//             lanes 16-31 -> M=lane-16, K = {kb+8..kb+15, kb+24..kb+31}.
__device__ __forceinline__ v16bf load_a_frag(const __bf16* base, int strideK,
                                             int kbase, int lane) {
    const int m = lane & 15;
    const __bf16* p = base + m * strideK + kbase + (lane >> 4) * 8;
    v8bf lo = *(const v8bf*)(p);
    v8bf hi = *(const v8bf*)(p + 16);
    v16bf r;
#pragma unroll
    for (int i = 0; i < 8; ++i) { r[i] = lo[i]; r[i + 8] = hi[i]; }
    return r;
}

// B fragment (32x16 K-major slab of column-major bf16 weights Wt[n][K]).
// ISA layout: lanes 0-15 -> N=lane, K = {kb..kb+7, kb+16..kb+23}; lanes 16-31 -> K = {kb+8..kb+15, kb+24..kb+31}.
__device__ __forceinline__ v16bf load_b_frag(const __bf16* __restrict__ Wt, int K,
                                             int ntile, int kbase, int lane) {
    const int n = ntile * 16 + (lane & 15);
    const __bf16* p = Wt + (size_t)n * K + kbase + (lane >> 4) * 8;
    v8bf lo = *(const v8bf*)(p);
    v8bf hi = *(const v8bf*)(p + 16);
    v16bf r;
#pragma unroll
    for (int i = 0; i < 8; ++i) { r[i] = lo[i]; r[i + 8] = hi[i]; }
    return r;
}

__device__ __forceinline__ void lds_fence() {
    asm volatile("" ::: "memory");
    __builtin_amdgcn_wave_barrier();
    asm volatile("" ::: "memory");
}

// ---------------- utility kernels ----------------

__global__ void convert_w_kernel(const float* __restrict__ W,
                                 __bf16* __restrict__ Wt, int K, int Nn) {
    int i = blockIdx.x * blockDim.x + threadIdx.x;
    if (i < K * Nn) {
        int k = i / Nn, n = i % Nn;
        Wt[(size_t)n * K + k] = (__bf16)W[i];
    }
}

__global__ void zero_kernel(float* a, long na, float* b, long nb, float* c, long nc) {
    long i = (long)blockIdx.x * blockDim.x + threadIdx.x;
    long stride = (long)gridDim.x * blockDim.x;
    for (long j = i; j < na; j += stride) a[j] = 0.0f;
    for (long j = i; j < nb; j += stride) b[j] = 0.0f;
    for (long j = i; j < nc; j += stride) c[j] = 0.0f;
}

__global__ void bn_finalize_kernel(const float* __restrict__ stats,
                                   const float* __restrict__ g,
                                   const float* __restrict__ be,
                                   float* __restrict__ bnp, float invCount) {
    int c = threadIdx.x;  // 128 threads
    float mean = stats[c] * invCount;
    float var = stats[c + 128] * invCount - mean * mean;
    float scale = g[c] * rsqrtf(var + 1e-5f);
    bnp[c] = scale;
    bnp[c + 128] = be[c] - mean * scale;
}

// ---------------- edge MLP pass 1: stats of Z1 ----------------

__global__ __launch_bounds__(256) void edge_pass1_kernel(
    const float* __restrict__ x, const float* __restrict__ ea,
    const int* __restrict__ erow,
    const __bf16* __restrict__ W1aT, const float* __restrict__ b1a,
    float* __restrict__ stats1, int ntiles) {
    __shared__ __bf16 Alds[8][16 * K1];
    __shared__ float statsL[256];
    const int tid = threadIdx.x, wave = tid >> 5, lane = tid & 31;
    statsL[tid] = 0.0f;
    __syncthreads();

    for (int t = blockIdx.x * 8 + wave; t < ntiles; t += gridDim.x * 8) {
        const int e0 = t * 16;
        // gather [x[row] || edge_attr] -> bf16 LDS tile (16 x 192)
        for (int idx = lane; idx < 16 * (K1 / 4); idx += 32) {
            int m = idx / (K1 / 4), c4 = idx % (K1 / 4);
            float4 v;
            if (c4 < XD / 4) {
                int r = erow[e0 + m];
                v = ((const float4*)(x + (size_t)r * XD))[c4];
            } else {
                v = ((const float4*)(ea + (size_t)(e0 + m) * ED))[c4 - XD / 4];
            }
            __bf16* dst = &Alds[wave][m * K1 + c4 * 4];
            dst[0] = (__bf16)v.x; dst[1] = (__bf16)v.y;
            dst[2] = (__bf16)v.z; dst[3] = (__bf16)v.w;
        }
        lds_fence();

#pragma unroll
        for (int nt = 0; nt < 8; ++nt) {
            const int c = nt * 16 + (lane & 15);
            const float bias = b1a[c];
            v8f acc;
#pragma unroll
            for (int i = 0; i < 8; ++i) acc[i] = bias;
#pragma unroll
            for (int kk = 0; kk < K1 / 32; ++kk) {
                v16bf a = load_a_frag(&Alds[wave][0], K1, kk * 32, lane);
                v16bf b = load_b_frag(W1aT, K1, nt, kk * 32, lane);
                acc = __builtin_amdgcn_wmma_f32_16x16x32_bf16(
                    false, a, false, b, (short)0, acc, false, false);
            }
            float s1 = 0.0f, s2 = 0.0f;
#pragma unroll
            for (int i = 0; i < 8; ++i) { float z = acc[i]; s1 += z; s2 += z * z; }
            atomicAdd(&statsL[c], s1);
            atomicAdd(&statsL[c + 128], s2);
        }
        lds_fence();
    }
    __syncthreads();
    atomicAdd(&stats1[tid], statsL[tid]);
}

// ---------------- edge MLP pass 2: selu(bn(Z1)) @ W1b + scatter ----------------

__global__ __launch_bounds__(256) void edge_pass2_kernel(
    const float* __restrict__ x, const float* __restrict__ ea,
    const int* __restrict__ erow, const int* __restrict__ ecol,
    const __bf16* __restrict__ W1aT, const float* __restrict__ b1a,
    const float* __restrict__ bn1,
    const __bf16* __restrict__ W1bT, const float* __restrict__ b1b,
    float* __restrict__ sbuf, float* __restrict__ cnt, int ntiles) {
    __shared__ __bf16 Alds[8][16 * K1];
    __shared__ int colL[8][16];
    const int tid = threadIdx.x, wave = tid >> 5, lane = tid & 31;

    for (int t = blockIdx.x * 8 + wave; t < ntiles; t += gridDim.x * 8) {
        const int e0 = t * 16;
        if (lane < 16) colL[wave][lane] = ecol[e0 + lane];
        for (int idx = lane; idx < 16 * (K1 / 4); idx += 32) {
            int m = idx / (K1 / 4), c4 = idx % (K1 / 4);
            float4 v;
            if (c4 < XD / 4) {
                int r = erow[e0 + m];
                v = ((const float4*)(x + (size_t)r * XD))[c4];
            } else {
                v = ((const float4*)(ea + (size_t)(e0 + m) * ED))[c4 - XD / 4];
            }
            __bf16* dst = &Alds[wave][m * K1 + c4 * 4];
            dst[0] = (__bf16)v.x; dst[1] = (__bf16)v.y;
            dst[2] = (__bf16)v.z; dst[3] = (__bf16)v.w;
        }
        lds_fence();

        // layer 1 for all 8 n-tiles, keep h in registers
        v8f h[8];
#pragma unroll
        for (int nt = 0; nt < 8; ++nt) {
            const int c = nt * 16 + (lane & 15);
            const float bias = b1a[c];
            v8f acc;
#pragma unroll
            for (int i = 0; i < 8; ++i) acc[i] = bias;
#pragma unroll
            for (int kk = 0; kk < K1 / 32; ++kk) {
                v16bf a = load_a_frag(&Alds[wave][0], K1, kk * 32, lane);
                v16bf b = load_b_frag(W1aT, K1, nt, kk * 32, lane);
                acc = __builtin_amdgcn_wmma_f32_16x16x32_bf16(
                    false, a, false, b, (short)0, acc, false, false);
            }
            const float scale = bn1[c], shift = bn1[c + 128];
#pragma unroll
            for (int i = 0; i < 8; ++i) h[nt][i] = selu_f(acc[i] * scale + shift);
        }
        lds_fence();

        // restage h (16x128 bf16) into the (now free) A buffer via D-layout
        __bf16* H = &Alds[wave][0];
#pragma unroll
        for (int nt = 0; nt < 8; ++nt) {
            const int c = nt * 16 + (lane & 15);
#pragma unroll
            for (int i = 0; i < 8; ++i) {
                const int m = i + ((lane >> 4) << 3);
                H[m * HD + c] = (__bf16)h[nt][i];
            }
        }
        lds_fence();

        // layer 2 + atomic scatter-add into s[col]
#pragma unroll
        for (int nt = 0; nt < 8; ++nt) {
            const int c = nt * 16 + (lane & 15);
            const float bias = b1b[c];
            v8f acc;
#pragma unroll
            for (int i = 0; i < 8; ++i) acc[i] = bias;
#pragma unroll
            for (int kk = 0; kk < HD / 32; ++kk) {
                v16bf a = load_a_frag(H, HD, kk * 32, lane);
                v16bf b = load_b_frag(W1bT, HD, nt, kk * 32, lane);
                acc = __builtin_amdgcn_wmma_f32_16x16x32_bf16(
                    false, a, false, b, (short)0, acc, false, false);
            }
#pragma unroll
            for (int i = 0; i < 8; ++i) {
                const int m = i + ((lane >> 4) << 3);
                const int r = colL[wave][m];
                atomicAdd(&sbuf[(size_t)r * HD + c], acc[i]);
            }
        }
        if (lane < 16) atomicAdd(&cnt[colL[wave][lane]], 1.0f);
        lds_fence();
    }
}

// ---------------- node MLP pass 1: stats of Z2 ----------------

__global__ __launch_bounds__(128) void node_pass1_kernel(
    const float* __restrict__ x, const float* __restrict__ sbuf,
    const float* __restrict__ cnt, const float* __restrict__ u,
    const int* __restrict__ batch,
    const __bf16* __restrict__ W2aT, const float* __restrict__ b2a,
    float* __restrict__ stats2, int ntiles) {
    __shared__ __bf16 Alds[4][16 * K2];
    __shared__ float statsL[256];
    const int tid = threadIdx.x, wave = tid >> 5, lane = tid & 31;
    statsL[tid] = 0.0f; statsL[tid + 128] = 0.0f;
    __syncthreads();

    for (int t = blockIdx.x * 4 + wave; t < ntiles; t += gridDim.x * 4) {
        const int n0 = t * 16;
        for (int idx = lane; idx < 16 * (K2 / 4); idx += 32) {
            int m = idx / (K2 / 4), c4 = idx % (K2 / 4);
            const int nid = n0 + m;
            float4 v;
            if (c4 < 32) {
                v = ((const float4*)(x + (size_t)nid * XD))[c4];
            } else if (c4 < 64) {
                v = ((const float4*)(sbuf + (size_t)nid * HD))[c4 - 32];
                const float inv = 1.0f / fmaxf(cnt[nid], 1.0f);
                v.x *= inv; v.y *= inv; v.z *= inv; v.w *= inv;
            } else {
                const int b = batch[nid];
                v = ((const float4*)(u + (size_t)b * UD))[c4 - 64];
            }
            __bf16* dst = &Alds[wave][m * K2 + c4 * 4];
            dst[0] = (__bf16)v.x; dst[1] = (__bf16)v.y;
            dst[2] = (__bf16)v.z; dst[3] = (__bf16)v.w;
        }
        lds_fence();

#pragma unroll
        for (int nt = 0; nt < 8; ++nt) {
            const int c = nt * 16 + (lane & 15);
            const float bias = b2a[c];
            v8f acc;
#pragma unroll
            for (int i = 0; i < 8; ++i) acc[i] = bias;
#pragma unroll
            for (int kk = 0; kk < K2 / 32; ++kk) {
                v16bf a = load_a_frag(&Alds[wave][0], K2, kk * 32, lane);
                v16bf b = load_b_frag(W2aT, K2, nt, kk * 32, lane);
                acc = __builtin_amdgcn_wmma_f32_16x16x32_bf16(
                    false, a, false, b, (short)0, acc, false, false);
            }
            float s1 = 0.0f, s2 = 0.0f;
#pragma unroll
            for (int i = 0; i < 8; ++i) { float z = acc[i]; s1 += z; s2 += z * z; }
            atomicAdd(&statsL[c], s1);
            atomicAdd(&statsL[c + 128], s2);
        }
        lds_fence();
    }
    __syncthreads();
    atomicAdd(&stats2[tid], statsL[tid]);
    atomicAdd(&stats2[tid + 128], statsL[tid + 128]);
}

// ---------------- node MLP pass 2: final output ----------------

__global__ __launch_bounds__(128) void node_pass2_kernel(
    const float* __restrict__ x, const float* __restrict__ sbuf,
    const float* __restrict__ cnt, const float* __restrict__ u,
    const int* __restrict__ batch,
    const __bf16* __restrict__ W2aT, const float* __restrict__ b2a,
    const float* __restrict__ bn2,
    const __bf16* __restrict__ W2bT, const float* __restrict__ b2b,
    float* __restrict__ out, int ntiles) {
    __shared__ __bf16 Alds[4][16 * K2];
    const int tid = threadIdx.x, wave = tid >> 5, lane = tid & 31;
    (void)tid;

    for (int t = blockIdx.x * 4 + wave; t < ntiles; t += gridDim.x * 4) {
        const int n0 = t * 16;
        for (int idx = lane; idx < 16 * (K2 / 4); idx += 32) {
            int m = idx / (K2 / 4), c4 = idx % (K2 / 4);
            const int nid = n0 + m;
            float4 v;
            if (c4 < 32) {
                v = ((const float4*)(x + (size_t)nid * XD))[c4];
            } else if (c4 < 64) {
                v = ((const float4*)(sbuf + (size_t)nid * HD))[c4 - 32];
                const float inv = 1.0f / fmaxf(cnt[nid], 1.0f);
                v.x *= inv; v.y *= inv; v.z *= inv; v.w *= inv;
            } else {
                const int b = batch[nid];
                v = ((const float4*)(u + (size_t)b * UD))[c4 - 64];
            }
            __bf16* dst = &Alds[wave][m * K2 + c4 * 4];
            dst[0] = (__bf16)v.x; dst[1] = (__bf16)v.y;
            dst[2] = (__bf16)v.z; dst[3] = (__bf16)v.w;
        }
        lds_fence();

        v8f h[8];
#pragma unroll
        for (int nt = 0; nt < 8; ++nt) {
            const int c = nt * 16 + (lane & 15);
            const float bias = b2a[c];
            v8f acc;
#pragma unroll
            for (int i = 0; i < 8; ++i) acc[i] = bias;
#pragma unroll
            for (int kk = 0; kk < K2 / 32; ++kk) {
                v16bf a = load_a_frag(&Alds[wave][0], K2, kk * 32, lane);
                v16bf b = load_b_frag(W2aT, K2, nt, kk * 32, lane);
                acc = __builtin_amdgcn_wmma_f32_16x16x32_bf16(
                    false, a, false, b, (short)0, acc, false, false);
            }
            const float scale = bn2[c], shift = bn2[c + 128];
#pragma unroll
            for (int i = 0; i < 8; ++i) h[nt][i] = selu_f(acc[i] * scale + shift);
        }
        lds_fence();

        __bf16* H = &Alds[wave][0];
#pragma unroll
        for (int nt = 0; nt < 8; ++nt) {
            const int c = nt * 16 + (lane & 15);
#pragma unroll
            for (int i = 0; i < 8; ++i) {
                const int m = i + ((lane >> 4) << 3);
                H[m * HD + c] = (__bf16)h[nt][i];
            }
        }
        lds_fence();

#pragma unroll
        for (int nt = 0; nt < 8; ++nt) {
            const int c = nt * 16 + (lane & 15);
            const float bias = b2b[c];
            v8f acc;
#pragma unroll
            for (int i = 0; i < 8; ++i) acc[i] = bias;
#pragma unroll
            for (int kk = 0; kk < HD / 32; ++kk) {
                v16bf a = load_a_frag(H, HD, kk * 32, lane);
                v16bf b = load_b_frag(W2bT, HD, nt, kk * 32, lane);
                acc = __builtin_amdgcn_wmma_f32_16x16x32_bf16(
                    false, a, false, b, (short)0, acc, false, false);
            }
#pragma unroll
            for (int i = 0; i < 8; ++i) {
                const int m = i + ((lane >> 4) << 3);
                out[(size_t)(n0 + m) * HD + c] = acc[i];
            }
        }
        lds_fence();
    }
}

static inline int imin_host(int a, int b) { return a < b ? a : b; }

extern "C" void kernel_launch(void* const* d_in, const int* in_sizes, int n_in,
                              void* d_out, int out_size, void* d_ws, size_t ws_size,
                              hipStream_t stream) {
    const float* x   = (const float*)d_in[0];
    const float* ea  = (const float*)d_in[1];
    const float* u   = (const float*)d_in[2];
    const float* W1a = (const float*)d_in[3];
    const float* b1a = (const float*)d_in[4];
    const float* g1  = (const float*)d_in[5];
    const float* be1 = (const float*)d_in[6];
    const float* W1b = (const float*)d_in[7];
    const float* b1b = (const float*)d_in[8];
    const float* W2a = (const float*)d_in[9];
    const float* b2a = (const float*)d_in[10];
    const float* g2  = (const float*)d_in[11];
    const float* be2 = (const float*)d_in[12];
    const float* W2b = (const float*)d_in[13];
    const float* b2b = (const float*)d_in[14];
    const int* eidx  = (const int*)d_in[15];
    const int* batch = (const int*)d_in[16];

    const int N = in_sizes[0] / XD;
    const int E = in_sizes[1] / ED;
    const int* erow = eidx;
    const int* ecol = eidx + E;

    char* ws = (char*)d_ws;
    __bf16* W1aT  = (__bf16*)(ws + OFF_W1AT);
    __bf16* W1bT  = (__bf16*)(ws + OFF_W1BT);
    __bf16* W2aT  = (__bf16*)(ws + OFF_W2AT);
    __bf16* W2bT  = (__bf16*)(ws + OFF_W2BT);
    float* stats1 = (float*)(ws + OFF_STATS1);
    float* bn1    = (float*)(ws + OFF_BN1);
    float* stats2 = (float*)(ws + OFF_STATS2);
    float* bn2    = (float*)(ws + OFF_BN2);
    float* sbuf   = (float*)(ws + OFF_S);
    float* cnt    = sbuf + (size_t)N * HD;

    // weights -> column-major bf16
    convert_w_kernel<<<(K1 * HD + 255) / 256, 256, 0, stream>>>(W1a, W1aT, K1, HD);
    convert_w_kernel<<<(HD * HD + 255) / 256, 256, 0, stream>>>(W1b, W1bT, HD, HD);
    convert_w_kernel<<<(K2 * HD + 255) / 256, 256, 0, stream>>>(W2a, W2aT, K2, HD);
    convert_w_kernel<<<(HD * HD + 255) / 256, 256, 0, stream>>>(W2b, W2bT, HD, HD);

    // zero stats (stats1..bn2 contiguous = 1024 f32), segment sums, counts
    zero_kernel<<<2048, 256, 0, stream>>>(stats1, 1024, sbuf, (long)N * HD, cnt, N);

    const int eT = E / 16;
    const int nT = N / 16;
    const int eGrid = imin_host((eT + 7) / 8, 2400);
    const int nGrid = imin_host((nT + 3) / 4, 1600);

    edge_pass1_kernel<<<eGrid, 256, 0, stream>>>(x, ea, erow, W1aT, b1a, stats1, eT);
    bn_finalize_kernel<<<1, 128, 0, stream>>>(stats1, g1, be1, bn1, 1.0f / (float)E);
    edge_pass2_kernel<<<eGrid, 256, 0, stream>>>(x, ea, erow, ecol, W1aT, b1a, bn1,
                                                 W1bT, b1b, sbuf, cnt, eT);
    node_pass1_kernel<<<nGrid, 128, 0, stream>>>(x, sbuf, cnt, u, batch, W2aT, b2a,
                                                 stats2, nT);
    bn_finalize_kernel<<<1, 128, 0, stream>>>(stats2, g2, be2, bn2, 1.0f / (float)N);
    node_pass2_kernel<<<nGrid, 128, 0, stream>>>(x, sbuf, cnt, u, batch, W2aT, b2a,
                                                 bn2, W2bT, b2b, (float*)d_out, nT);
}
